// IntentAwareSelfAttention_70291434766387
// MI455X (gfx1250) — matrix operand order repaired and checked by reading, not compiled, not run on hardware
//
#include <hip/hip_runtime.h>

// ---------------- problem constants ----------------
#define Ecst  1024
#define Hcst  16
#define Dcst  64
#define Bcst  4
#define Scst  2048
#define DIcst 64
#define Kcst  32
#define MROWS (Bcst * Scst)   // 8192

typedef __attribute__((ext_vector_type(16))) __bf16 v16bf;
typedef __attribute__((ext_vector_type(8)))  __bf16 bf8;
typedef __attribute__((ext_vector_type(8)))  float  v8f;

union V16U { v16bf v; bf8 h[2]; };

static __device__ inline v8f wmma_bf16(v16bf a, v16bf b, v8f c) {
  return __builtin_amdgcn_wmma_f32_16x16x32_bf16(false, a, false, b, (short)0, c, false, false);
}

// A-fragment: 16x32 bf16, row-major source with leading dim ld, col offset c0.
// lane<16: row=lane, K = {c0+0..7, c0+16..23}; lane>=16: row=lane-16, K = {c0+8..15, c0+24..31}
static __device__ inline v16bf load_frag_a(const __bf16* base, int ld, int c0, int lane) {
  int hf  = lane >> 4;
  int row = lane & 15;
  const __bf16* rp = base + (size_t)row * ld + c0 + hf * 8;
  V16U u;
  u.h[0] = *(const bf8*)(rp);
  u.h[1] = *(const bf8*)(rp + 16);
  return u.v;
}

// B-fragment: 32x16, where B[K][N] = Mrow[N][c0+K] for row-major Mrow with leading dim ld.
// lane: N = lane&15, K = {c0+8*(lane>>4)+0..7, c0+16+8*(lane>>4)+0..7}
static __device__ inline v16bf load_frag_b(const __bf16* base, int ld, int c0, int lane) {
  int hf = lane >> 4;
  int n  = lane & 15;
  const __bf16* rp = base + (size_t)n * ld + c0 + hf * 8;
  V16U u;
  u.h[0] = *(const bf8*)(rp);
  u.h[1] = *(const bf8*)(rp + 16);
  return u.v;
}

// ---------------- kernel: fp32 -> bf16 convert ----------------
__global__ void cvt_f32_bf16_kernel(const float* __restrict__ src, __bf16* __restrict__ dst, int n) {
  int idx = blockIdx.x * blockDim.x + threadIdx.x;
  int stride = gridDim.x * blockDim.x;
  for (; idx < n; idx += stride) dst[idx] = (__bf16)src[idx];
}

// ---------------- kernel: per-(b,s) stats ----------------
__global__ void stats_kernel(const float* __restrict__ mu, const float* __restrict__ sigma,
                             const float* __restrict__ explore, const float* __restrict__ exploit,
                             const unsigned char* __restrict__ msk,
                             float* __restrict__ sq, float* __restrict__ unc, float* __restrict__ tinv) {
  int idx = blockIdx.x * blockDim.x + threadIdx.x;
  if (idx >= MROWS) return;
  const float* mp = mu + (size_t)idx * DIcst;
  const float* sp = sigma + (size_t)idx * DIcst;
  float s2 = 0.f, su = 0.f;
#pragma unroll 8
  for (int i = 0; i < DIcst; ++i) { float m = mp[i]; s2 += m * m; su += sp[i]; }
  sq[idx]  = s2;
  unc[idx] = su * (1.0f / DIcst);
  float t = 1.0f + 0.5f * explore[idx] - 0.5f * exploit[idx];
  t = fminf(fmaxf(t, 0.5f), 2.0f);
  if (msk[idx]) t = 1.0f;
  tinv[idx] = 1.0f / t;
}

// ---------------- kernel: bf16 GEMM  C = A(MxK) * Bw(NxK)^T + bias ----------------
// Software-pipelined: fragment ping-pong so WMMA(i) overlaps loads(i+1).
// mode 0: outB row-major bf16 (ld N)
// mode 1: outB = V transposed, layout (B,H,D,S)
// mode 2: outF fp32 row-major + zero rows where rowmask[m] != 0
__global__ void gemm_bf16_kernel(const __bf16* __restrict__ A, const __bf16* __restrict__ Bw,
                                 const float* __restrict__ bias,
                                 int M, int N, int Kd, int mode,
                                 __bf16* __restrict__ outB, float* __restrict__ outF,
                                 const unsigned char* __restrict__ rowmask) {
  int lane = threadIdx.x & 31;
  int w    = threadIdx.x >> 5;
  int row0 = blockIdx.y * 64 + w * 16;
  int col0 = blockIdx.x * 64;
  int hf   = lane >> 4;
  int n    = lane & 15;

  v8f acc[4];
#pragma unroll
  for (int t = 0; t < 4; ++t)
#pragma unroll
    for (int r = 0; r < 8; ++r) acc[t][r] = 0.f;

  const __bf16* Abase = A + (size_t)row0 * Kd;
  const int steps = Kd >> 5;             // Kd is a power of two multiple of 32

  // prologue: fragments for K-step 0
  v16bf aC = load_frag_a(Abase, Kd, 0, lane);
  v16bf bC[4];
#pragma unroll
  for (int t = 0; t < 4; ++t)
    bC[t] = load_frag_b(Bw + (size_t)(col0 + t * 16) * Kd, Kd, 0, lane);

#pragma unroll 2
  for (int i = 0; i < steps; ++i) {
    int kn = ((i + 1) & (steps - 1)) << 5;      // wrap-around dummy load on last step
    int kp = ((i + 2) & (steps - 1)) << 5;
    __builtin_prefetch((const void*)(Abase + (size_t)(lane & 15) * Kd + kp), 0, 0);

    v16bf aN = load_frag_a(Abase, Kd, kn, lane);
    v16bf bN[4];
#pragma unroll
    for (int t = 0; t < 4; ++t)
      bN[t] = load_frag_b(Bw + (size_t)(col0 + t * 16) * Kd, Kd, kn, lane);

#pragma unroll
    for (int t = 0; t < 4; ++t)
      acc[t] = wmma_bf16(aC, bC[t], acc[t]);

    aC = aN;
#pragma unroll
    for (int t = 0; t < 4; ++t) bC[t] = bN[t];
  }

#pragma unroll
  for (int t = 0; t < 4; ++t) {
    int nn = col0 + t * 16 + n;
    float bv = bias ? bias[nn] : 0.f;
#pragma unroll
    for (int r = 0; r < 8; ++r) {
      int m = row0 + r + 8 * hf;
      float v = acc[t][r] + bv;
      if (mode == 0) {
        outB[(size_t)m * N + nn] = (__bf16)v;
      } else if (mode == 1) {
        int b = m / Scst, s = m - b * Scst;
        int h = nn >> 6, d = nn & 63;
        outB[((size_t)((b * Hcst + h) * Dcst + d)) * Scst + s] = (__bf16)v;
      } else {
        if (rowmask[m]) v = 0.f;
        outF[(size_t)m * N + nn] = v;
      }
    }
  }
}

// ---------------- kernel: bias precompute (bf16 out) ----------------
// bias[b,q,k] = 0.5*assign_dot - 0.5*((sq[q]+sq[k]-2*mu_dot)/DI) - 0.5*unc[k]
__global__ void bias_kernel(const __bf16* __restrict__ assignB, const __bf16* __restrict__ muB,
                            const float* __restrict__ sq, const float* __restrict__ unc,
                            __bf16* __restrict__ biasB) {
  int lane = threadIdx.x & 31;
  int w    = threadIdx.x >> 5;
  int b    = blockIdx.z;
  int q0   = blockIdx.y * 16;
  int k0   = blockIdx.x * 64 + w * 16;
  int hf   = lane >> 4;
  int n    = lane & 15;

  v8f da, dm;
#pragma unroll
  for (int r = 0; r < 8; ++r) { da[r] = 0.f; dm[r] = 0.f; }

  const __bf16* aq = assignB + (size_t)(b * Scst + q0) * Kcst;
  const __bf16* ak = assignB + (size_t)(b * Scst + k0) * Kcst;
  da = wmma_bf16(load_frag_a(aq, Kcst, 0, lane), load_frag_b(ak, Kcst, 0, lane), da);

  const __bf16* mq = muB + (size_t)(b * Scst + q0) * DIcst;
  const __bf16* mk = muB + (size_t)(b * Scst + k0) * DIcst;
  dm = wmma_bf16(load_frag_a(mq, DIcst, 0,  lane), load_frag_b(mk, DIcst, 0,  lane), dm);
  dm = wmma_bf16(load_frag_a(mq, DIcst, 32, lane), load_frag_b(mk, DIcst, 32, lane), dm);

  int kk = k0 + n;
  float sqk  = sq[b * Scst + kk];
  float unck = unc[b * Scst + kk];
#pragma unroll
  for (int r = 0; r < 8; ++r) {
    int q = q0 + r + 8 * hf;
    float sqq = sq[b * Scst + q];
    float mu_dist = (sqq + sqk - 2.0f * dm[r]) * (1.0f / DIcst);
    float v = 0.5f * da[r] - 0.5f * mu_dist - 0.5f * unck;
    biasB[((size_t)(b * Scst + q)) * Scst + kk] = (__bf16)v;
  }
}

// ---------------- kernel: flash attention ----------------
// grid: (S/64, B*H), block 128 (4 waves); wave handles 16 q-rows, loops keys in steps of 32.
// K-fragments are ping-ponged; V-fragments are issued before the softmax VALU section so the
// XDL/VALU work overlaps the in-flight loads.
__global__ void flash_attn_kernel(const __bf16* __restrict__ Qb, const __bf16* __restrict__ Kb,
                                  const __bf16* __restrict__ Vt, const __bf16* __restrict__ biasB,
                                  const unsigned char* __restrict__ msk, const float* __restrict__ tinv,
                                  __bf16* __restrict__ attnB) {
  __shared__ __bf16 pLds[4][16 * 32];

  int lane = threadIdx.x & 31;
  int w    = threadIdx.x >> 5;
  int bh   = blockIdx.y;
  int b    = bh / Hcst;
  int h    = bh - b * Hcst;
  int m0   = blockIdx.x * 64 + w * 16;
  int hf   = lane >> 4;
  int n    = lane & 15;

  // Q fragments (held for the whole loop)
  const __bf16* qbase = Qb + (size_t)(b * Scst + m0) * Ecst + h * Dcst;
  v16bf qA0 = load_frag_a(qbase, Ecst, 0,  lane);
  v16bf qA1 = load_frag_a(qbase, Ecst, 32, lane);

  float mrow[8], lrow[8], trow[8];
  v8f oAcc[4];
#pragma unroll
  for (int r = 0; r < 8; ++r) {
    mrow[r] = -1e30f; lrow[r] = 0.f;
    trow[r] = tinv[b * Scst + m0 + r + 8 * hf];
  }
#pragma unroll
  for (int t = 0; t < 4; ++t)
#pragma unroll
    for (int r = 0; r < 8; ++r) oAcc[t][r] = 0.f;

  const float scale = 0.125f;  // 1/sqrt(64)
  const __bf16* Kbh = Kb + (size_t)b * Scst * Ecst + h * Dcst;

  // prologue: K fragments for k0 = 0
  v16bf kf00 = load_frag_b(Kbh,              Ecst, 0,  lane);
  v16bf kf01 = load_frag_b(Kbh,              Ecst, 32, lane);
  v16bf kf10 = load_frag_b(Kbh + 16 * Ecst,  Ecst, 0,  lane);
  v16bf kf11 = load_frag_b(Kbh + 16 * Ecst,  Ecst, 32, lane);

#pragma unroll 2
  for (int k0 = 0; k0 < Scst; k0 += 32) {
    // V fragments for this step: issue early, consumed only after softmax
    v16bf vf[4];
#pragma unroll
    for (int t = 0; t < 4; ++t) {
      const __bf16* vbp = Vt + ((size_t)(bh * Dcst + t * 16)) * Scst + k0;
      vf[t] = load_frag_b(vbp, Scst, 0, lane);
    }

    // QK^T on current K fragments
    v8f sacc[2];
#pragma unroll
    for (int r = 0; r < 8; ++r) { sacc[0][r] = 0.f; sacc[1][r] = 0.f; }
    sacc[0] = wmma_bf16(qA0, kf00, sacc[0]);
    sacc[0] = wmma_bf16(qA1, kf01, sacc[0]);
    sacc[1] = wmma_bf16(qA0, kf10, sacc[1]);
    sacc[1] = wmma_bf16(qA1, kf11, sacc[1]);

    // next-step K fragments (ping-pong; wrap-around dummy on last step)
    int knx = (k0 + 32) & (Scst - 1);
    int kpf = (k0 + 64) & (Scst - 1);
    __builtin_prefetch((const void*)(Kbh + (size_t)(kpf + n) * Ecst + 16 * hf), 0, 0);
    v16bf kn00 = load_frag_b(Kbh + (size_t)knx * Ecst,        Ecst, 0,  lane);
    v16bf kn01 = load_frag_b(Kbh + (size_t)knx * Ecst,        Ecst, 32, lane);
    v16bf kn10 = load_frag_b(Kbh + (size_t)(knx + 16) * Ecst, Ecst, 0,  lane);
    v16bf kn11 = load_frag_b(Kbh + (size_t)(knx + 16) * Ecst, Ecst, 32, lane);

    // epilogue on logits: bias, mask, temperature
    float sv[2][8];
    float mmul[2];
#pragma unroll
    for (int kt = 0; kt < 2; ++kt) {
      int kbase = k0 + kt * 16;
      mmul[kt] = msk[b * Scst + kbase + n] ? 0.f : 1.f;
      const __bf16* bp = biasB + (size_t)(b * Scst + m0) * Scst + kbase + n;
#pragma unroll
      for (int r = 0; r < 8; ++r) {
        float bb = (float)bp[(size_t)(r + 8 * hf) * Scst];
        float s  = (sacc[kt][r] * scale + bb) * trow[r];
        sv[kt][r] = (mmul[kt] > 0.f) ? s : -1e30f;
      }
    }

    // online softmax per row (row M = r + 8*hf lives in one 16-lane half)
#pragma unroll
    for (int r = 0; r < 8; ++r) {
      float tmax = fmaxf(sv[0][r], sv[1][r]);
#pragma unroll
      for (int sh = 1; sh <= 8; sh <<= 1) tmax = fmaxf(tmax, __shfl_xor(tmax, sh, 32));
      float newm = fmaxf(mrow[r], tmax);
      float al = __expf(mrow[r] - newm);
      mrow[r] = newm;
      float p0 = mmul[0] * __expf(sv[0][r] - newm);
      float p1 = mmul[1] * __expf(sv[1][r] - newm);
      float ts = p0 + p1;
#pragma unroll
      for (int sh = 1; sh <= 8; sh <<= 1) ts += __shfl_xor(ts, sh, 32);
      lrow[r] = lrow[r] * al + ts;
#pragma unroll
      for (int t = 0; t < 4; ++t) oAcc[t][r] *= al;
      int prow = r + 8 * hf;
      pLds[w][prow * 32 + n]      = (__bf16)p0;
      pLds[w][prow * 32 + 16 + n] = (__bf16)p1;
    }

    // reshuffle P (C layout -> A layout) through LDS, then P @ V
    v16bf pA = load_frag_a(&pLds[w][0], 32, 0, lane);
#pragma unroll
    for (int t = 0; t < 4; ++t)
      oAcc[t] = wmma_bf16(pA, vf[t], oAcc[t]);

    kf00 = kn00; kf01 = kn01; kf10 = kn10; kf11 = kn11;
  }

#pragma unroll
  for (int t = 0; t < 4; ++t)
#pragma unroll
    for (int r = 0; r < 8; ++r) {
      float o = oAcc[t][r] / fmaxf(lrow[r], 1e-8f);
      int q = m0 + r + 8 * hf;
      attnB[(size_t)(b * Scst + q) * Ecst + h * Dcst + t * 16 + n] = (__bf16)o;
    }
}

// ---------------- host side ----------------
static inline size_t align256(size_t x) { return (x + 255) & ~(size_t)255; }

extern "C" void kernel_launch(void* const* d_in, const int* in_sizes, int n_in,
                              void* d_out, int out_size, void* d_ws, size_t ws_size,
                              hipStream_t stream) {
  const float* his     = (const float*)d_in[0];
  const unsigned char* msk = (const unsigned char*)d_in[1];
  const float* mu      = (const float*)d_in[2];
  const float* sigma   = (const float*)d_in[3];
  const float* assign  = (const float*)d_in[4];
  const float* explore = (const float*)d_in[5];
  const float* exploit = (const float*)d_in[6];
  const float* Wq = (const float*)d_in[7];
  const float* bq = (const float*)d_in[8];
  const float* Wk = (const float*)d_in[9];
  const float* bk = (const float*)d_in[10];
  const float* Wv = (const float*)d_in[11];
  const float* bv = (const float*)d_in[12];
  const float* Wo = (const float*)d_in[13];
  const float* bo = (const float*)d_in[14];
  float* outF = (float*)d_out;

  char* p = (char*)d_ws;
  auto take = [&](size_t bytes) { char* r = p; p += align256(bytes); return r; };

  __bf16* hisB    = (__bf16*)take((size_t)MROWS * Ecst * 2);
  __bf16* WqB     = (__bf16*)take((size_t)Ecst * Ecst * 2);
  __bf16* WkB     = (__bf16*)take((size_t)Ecst * Ecst * 2);
  __bf16* WvB     = (__bf16*)take((size_t)Ecst * Ecst * 2);
  __bf16* WoB     = (__bf16*)take((size_t)Ecst * Ecst * 2);
  __bf16* assignB = (__bf16*)take((size_t)MROWS * Kcst * 2);
  __bf16* muB     = (__bf16*)take((size_t)MROWS * DIcst * 2);
  float*  sq      = (float*)take((size_t)MROWS * 4);
  float*  unc     = (float*)take((size_t)MROWS * 4);
  float*  tinv    = (float*)take((size_t)MROWS * 4);
  __bf16* Qb      = (__bf16*)take((size_t)MROWS * Ecst * 2);
  __bf16* Kb      = (__bf16*)take((size_t)MROWS * Ecst * 2);
  __bf16* Vt      = (__bf16*)take((size_t)MROWS * Ecst * 2);
  __bf16* biasB   = (__bf16*)take((size_t)Bcst * Scst * Scst * 2);
  __bf16* attnB   = (__bf16*)take((size_t)MROWS * Ecst * 2);

  auto cvt = [&](const float* s, __bf16* d, int n) {
    int blocks = (n + 1023) / 1024; if (blocks > 16384) blocks = 16384;
    cvt_f32_bf16_kernel<<<blocks, 256, 0, stream>>>(s, d, n);
  };
  cvt(his,    hisB,    MROWS * Ecst);
  cvt(Wq,     WqB,     Ecst * Ecst);
  cvt(Wk,     WkB,     Ecst * Ecst);
  cvt(Wv,     WvB,     Ecst * Ecst);
  cvt(Wo,     WoB,     Ecst * Ecst);
  cvt(assign, assignB, MROWS * Kcst);
  cvt(mu,     muB,     MROWS * DIcst);

  stats_kernel<<<(MROWS + 255) / 256, 256, 0, stream>>>(mu, sigma, explore, exploit, msk, sq, unc, tinv);

  dim3 ggrid(Ecst / 64, MROWS / 64);
  gemm_bf16_kernel<<<ggrid, 128, 0, stream>>>(hisB, WqB, bq, MROWS, Ecst, Ecst, 0, Qb, nullptr, nullptr);
  gemm_bf16_kernel<<<ggrid, 128, 0, stream>>>(hisB, WkB, bk, MROWS, Ecst, Ecst, 0, Kb, nullptr, nullptr);
  gemm_bf16_kernel<<<ggrid, 128, 0, stream>>>(hisB, WvB, bv, MROWS, Ecst, Ecst, 1, Vt, nullptr, nullptr);

  dim3 bgrid(Scst / 64, Scst / 16, Bcst);
  bias_kernel<<<bgrid, 128, 0, stream>>>(assignB, muB, sq, unc, biasB);

  dim3 fgrid(Scst / 64, Bcst * Hcst);
  flash_attn_kernel<<<fgrid, 128, 0, stream>>>(Qb, Kb, Vt, biasB, msk, tinv, attnB);

  gemm_bf16_kernel<<<ggrid, 128, 0, stream>>>(attnB, WoB, bo, MROWS, Ecst, Ecst, 2, nullptr, outF, msk);
}
